// SelfAttentionBlock_60473139527892
// MI455X (gfx1250) — matrix operand order repaired and checked by reading, not compiled, not run on hardware
//
#include <hip/hip_runtime.h>
#include <hip/hip_bf16.h>

// ---------------------------------------------------------------------------
// SelfAttentionBlock for MI455X (gfx1250), wave32 + WMMA flash attention.
// B=4, C=64, N=4096.  q/k/v projected to f16 (1/sqrt(C) folded into q),
// attention computed as S^T = K*Q^T tiles with v_wmma_f32_16x16x32_f16,
// online softmax (stats live in the lane dimension thanks to the transpose),
// then O += P*V with the P fragment built lane-locally from the S^T D-frags.
// K/V chunks (64 j-positions) are staged into double-buffered LDS with
// global_load_async_to_lds_b128 (ASYNCcnt), one barrier per chunk.
// ---------------------------------------------------------------------------

typedef __attribute__((ext_vector_type(16))) _Float16 v16h;
typedef __attribute__((ext_vector_type(8)))  _Float16 v8h;
typedef __attribute__((ext_vector_type(8)))  float    v8f;
typedef __attribute__((ext_vector_type(4)))  unsigned int v4u;
typedef int v4i __attribute__((vector_size(16)));  // GCC vector: matches builtin

constexpr int BB = 4;
constexpr int CC = 64;
constexpr int NN = 4096;

#define ASG __attribute__((address_space(1)))
#define ASL __attribute__((address_space(3)))

#if defined(__has_builtin)
#if __has_builtin(__builtin_amdgcn_global_load_async_to_lds_b128)
#define USE_ASYNC_LDS 1
#endif
#endif
#ifndef USE_ASYNC_LDS
#define USE_ASYNC_LDS 0
#endif

#if defined(__has_builtin)
#if __has_builtin(__builtin_amdgcn_s_wait_asynccnt)
#define WAIT_ASYNC() __builtin_amdgcn_s_wait_asynccnt(0)
#else
#define WAIT_ASYNC() asm volatile("s_wait_asynccnt 0x0" ::: "memory")
#endif
#endif

__device__ __forceinline__ v16h cat16(v8h lo, v8h hi) {
  v16h r;
#pragma unroll
  for (int i = 0; i < 8; ++i) { r[i] = lo[i]; r[i + 8] = hi[i]; }
  return r;
}

// ---------------------------------------------------------------------------
// Projection: q = (Wq x + bq) * 0.125 (f16, [b][n][c])
//             k = (Wk x + bk)          (f16, [b][n][c])
//             vT = (Wv x + bv)         (f16, [b][c][n])  (transposed for P*V)
// ---------------------------------------------------------------------------
__global__ __launch_bounds__(256) void proj_qkv(
    const float* __restrict__ x,
    const float* __restrict__ Wq, const float* __restrict__ bq,
    const float* __restrict__ Wk, const float* __restrict__ bk,
    const float* __restrict__ Wv, const float* __restrict__ bv,
    _Float16* __restrict__ q16, _Float16* __restrict__ k16,
    _Float16* __restrict__ vT16) {
  const int b  = blockIdx.x / (NN / 64);
  const int n0 = (blockIdx.x % (NN / 64)) * 64;
  const int t  = threadIdx.x;

  __shared__ float sx[64 * 64];  // [c][nl]
  for (int idx = t; idx < 64 * 64; idx += 256) {
    const int c = idx >> 6, nl = idx & 63;
    sx[idx] = x[((size_t)(b * CC + c)) * NN + n0 + nl];
  }
  __syncthreads();

  const int o  = t & 63;
  const int ng = t >> 6;  // 4 position-groups
#pragma unroll 1
  for (int nl = ng; nl < 64; nl += 4) {
    float aq = 0.f, ak = 0.f, av = 0.f;
#pragma unroll
    for (int c = 0; c < 64; ++c) {
      const float xv = sx[c * 64 + nl];
      aq = fmaf(Wq[o * 64 + c], xv, aq);
      ak = fmaf(Wk[o * 64 + c], xv, ak);
      av = fmaf(Wv[o * 64 + c], xv, av);
    }
    aq += bq[o]; ak += bk[o]; av += bv[o];
    const size_t row = (size_t)(b * NN + n0 + nl) * 64;
    q16[row + o]  = (_Float16)(aq * 0.125f);  // fold 1/sqrt(C)
    k16[row + o]  = (_Float16)ak;
    vT16[((size_t)(b * CC + o)) * NN + n0 + nl] = (_Float16)av;
  }
}

// ---------------------------------------------------------------------------
// Flash attention. 8 waves / block, each wave owns a 16-row i-tile.
// 64-j chunks, double-buffered LDS staging, one barrier per chunk.
// ---------------------------------------------------------------------------
constexpr int KPAD = 72;  // halves per sK row (144B, 16B aligned, bank-spread)
constexpr int VPAD = 72;  // halves per sV row
constexpr int KBUF = 64 * KPAD;
constexpr int VBUF = 64 * VPAD;

__global__ __launch_bounds__(256) void flash_attn(
    const _Float16* __restrict__ q16, const _Float16* __restrict__ k16,
    const _Float16* __restrict__ vT16, const float* __restrict__ x,
    const float* __restrict__ gamma, float* __restrict__ out) {
  constexpr int TILES_PER_B = NN / 128;  // 8 waves * 16 rows
  constexpr int NCHUNK = NN / 64;
  const int b    = blockIdx.x / TILES_PER_B;
  const int it   = blockIdx.x % TILES_PER_B;
  const int t    = threadIdx.x;
  const int wave = t >> 5;
  const int lane = t & 31;
  const int half = lane >> 4;
  const int l15  = lane & 15;
  const int i0   = it * 128 + wave * 16;

  __shared__ __align__(16) _Float16 sK[2 * KBUF];  // [buf][j_local][c]
  __shared__ __align__(16) _Float16 sV[2 * VBUF];  // [buf][c][j_local]

  // Q B-fragments (col = i, K = c), loaded once per wave. B-layout:
  // lanes 0-15 hold K {0..7,16..23}, lanes 16-31 hold K {8..15,24..31} (matches A).
  const _Float16* qrow = q16 + (size_t)(b * NN + i0 + l15) * 64;
  const v16h bq0 = cat16(*(const v8h*)(qrow + half * 8), *(const v8h*)(qrow + 16 + half * 8));       // c 0..31
  const v16h bq1 = cat16(*(const v8h*)(qrow + 32 + half * 8), *(const v8h*)(qrow + 48 + half * 8));  // c 32..63

  v8f oacc[4];
#pragma unroll
  for (int ct = 0; ct < 4; ++ct)
#pragma unroll
    for (int r = 0; r < 8; ++r) oacc[ct][r] = 0.f;
  float m_run = -3.0e38f, l_run = 0.f;

  // staging coordinates: 2 x (16B of K, 16B of V) per thread per chunk
  const int r0  = t >> 3;   // 0..31
  const int seg = t & 7;    // 8 x 16B segments per 64-half row

  auto stage = [&](int buf, int j0) {
#pragma unroll
    for (int h = 0; h < 2; ++h) {
      const int row = r0 + h * 32;  // 0..63
      const _Float16* gk = k16 + (size_t)(b * NN + j0 + row) * 64 + seg * 8;
      const _Float16* gv = vT16 + (size_t)(b * CC + row) * NN + j0 + seg * 8;
      _Float16* lk = sK + buf * KBUF + row * KPAD + seg * 8;
      _Float16* lv = sV + buf * VBUF + row * VPAD + seg * 8;
#if USE_ASYNC_LDS
      __builtin_amdgcn_global_load_async_to_lds_b128((ASG v4i*)gk, (ASL v4i*)lk, 0, 0);
      __builtin_amdgcn_global_load_async_to_lds_b128((ASG v4i*)gv, (ASL v4i*)lv, 0, 0);
#else
      *(v4u*)lk = *(const v4u*)gk;
      *(v4u*)lv = *(const v4u*)gv;
#endif
    }
  };

  stage(0, 0);

#pragma unroll 1
  for (int cch = 0; cch < NCHUNK; ++cch) {
#if USE_ASYNC_LDS
    WAIT_ASYNC();  // my async stores to LDS have landed
#endif
    __syncthreads();  // publishes chunk cch; retires reads of the other buffer
    if (cch + 1 < NCHUNK) stage((cch + 1) & 1, (cch + 1) * 64);

    const _Float16* bK = sK + (cch & 1) * KBUF;
    const _Float16* bV = sV + (cch & 1) * VBUF;

    // ---- S^T tiles d[jt] = K[j0+16jt .. +16) * Q^T over K=c=64 ----
    // A-layout (16-bit, 16x32): lanes0-15 K∈{0..7,16..23}, lanes16-31 K∈{8..15,24..31}
    v8f d[4];
    const v8f zero = {0.f, 0.f, 0.f, 0.f, 0.f, 0.f, 0.f, 0.f};
#pragma unroll
    for (int jt = 0; jt < 4; ++jt) {
      const _Float16* kr = &bK[(jt * 16 + l15) * KPAD + half * 8];
      const v16h a0 = cat16(*(const v8h*)kr, *(const v8h*)(kr + 16));
      const v16h a1 = cat16(*(const v8h*)(kr + 32), *(const v8h*)(kr + 48));
      d[jt] = __builtin_amdgcn_wmma_f32_16x16x32_f16(false, a0, false, bq0,
                                                     (short)0, zero, false, false);
      d[jt] = __builtin_amdgcn_wmma_f32_16x16x32_f16(false, a1, false, bq1,
                                                     (short)0, d[jt], false, false);
    }

    // ---- online softmax; column of S^T (= row i) lives in this lane ----
    float cm = -3.0e38f;
#pragma unroll
    for (int jt = 0; jt < 4; ++jt)
#pragma unroll
      for (int r = 0; r < 8; ++r) cm = fmaxf(cm, d[jt][r]);
    cm = fmaxf(cm, __shfl_xor(cm, 16, 32));
    const float nm    = fmaxf(m_run, cm);
    const float alpha = __expf(m_run - nm);
    float p[4][8], psum = 0.f;
#pragma unroll
    for (int jt = 0; jt < 4; ++jt)
#pragma unroll
      for (int r = 0; r < 8; ++r) {
        p[jt][r] = __expf(d[jt][r] - nm);
        psum += p[jt][r];
      }
    psum += __shfl_xor(psum, 16, 32);
    l_run = l_run * alpha + psum;
    m_run = nm;

    // rescale O: row m = r + 8*half; stats for row m sit in lane m
#pragma unroll
    for (int r = 0; r < 8; ++r) {
      const float fr = __shfl(alpha, r + 8 * half, 32);
      oacc[0][r] *= fr; oacc[1][r] *= fr; oacc[2][r] *= fr; oacc[3][r] *= fr;
    }

    // ---- P A-fragments: lane-local repack of d[] (the S^T payoff) ----
    v16h ap0, ap1;
#pragma unroll
    for (int r = 0; r < 8; ++r) {
      ap0[r]     = (_Float16)p[0][r];  // K = half*8 + r
      ap0[r + 8] = (_Float16)p[1][r];  // K = 16 + half*8 + r
      ap1[r]     = (_Float16)p[2][r];  // K = 32 + half*8 + r
      ap1[r + 8] = (_Float16)p[3][r];  // K = 48 + half*8 + r
    }

    // ---- O += P * V over 4 c-tiles, two K=32 j-windows ----
#pragma unroll
    for (int ct = 0; ct < 4; ++ct) {
      const _Float16* vr = &bV[(ct * 16 + l15) * VPAD + half * 8];
      const v16h bv0 = cat16(*(const v8h*)vr, *(const v8h*)(vr + 16));
      const v16h bv1 = cat16(*(const v8h*)(vr + 32), *(const v8h*)(vr + 48));
      oacc[ct] = __builtin_amdgcn_wmma_f32_16x16x32_f16(false, ap0, false, bv0,
                                                        (short)0, oacc[ct],
                                                        false, false);
      oacc[ct] = __builtin_amdgcn_wmma_f32_16x16x32_f16(false, ap1, false, bv1,
                                                        (short)0, oacc[ct],
                                                        false, false);
    }
  }

  // ---- epilogue: out = gamma * O/l + x ----
  const float g    = gamma[0];
  const float linv = 1.0f / l_run;
#pragma unroll
  for (int r = 0; r < 8; ++r) {
    const float fr = __shfl(linv, r + 8 * half, 32);
    const int i = i0 + r + 8 * half;
#pragma unroll
    for (int ct = 0; ct < 4; ++ct) {
      const int c = ct * 16 + l15;
      const size_t idx = (size_t)(b * CC + c) * NN + i;
      out[idx] = g * (oacc[ct][r] * fr) + x[idx];
    }
  }
}

// ---------------------------------------------------------------------------
extern "C" void kernel_launch(void* const* d_in, const int* in_sizes, int n_in,
                              void* d_out, int out_size, void* d_ws, size_t ws_size,
                              hipStream_t stream) {
  (void)in_sizes; (void)n_in; (void)out_size; (void)ws_size;
  const float* x     = (const float*)d_in[0];
  const float* Wq    = (const float*)d_in[1];
  const float* bq    = (const float*)d_in[2];
  const float* Wk    = (const float*)d_in[3];
  const float* bk    = (const float*)d_in[4];
  const float* Wv    = (const float*)d_in[5];
  const float* bv    = (const float*)d_in[6];
  const float* gamma = (const float*)d_in[7];
  float* out = (float*)d_out;

  _Float16* q16  = (_Float16*)d_ws;                       // 2 MB
  _Float16* k16  = q16 + (size_t)BB * NN * CC;            // 2 MB
  _Float16* vT16 = k16 + (size_t)BB * NN * CC;            // 2 MB

  proj_qkv<<<BB * (NN / 64), 256, 0, stream>>>(x, Wq, bq, Wk, bk, Wv, bv,
                                               q16, k16, vT16);
  flash_attn<<<BB * (NN / 128), 256, 0, stream>>>(q16, k16, vT16, x, gamma, out);
}
